// MixingBlock_10144712753909
// MI455X (gfx1250) — matrix operand order repaired and checked by reading, not compiled, not run on hardware
//
#include <hip/hip_runtime.h>
#include <math.h>

typedef _Float16 f16;
typedef __attribute__((ext_vector_type(8)))  _Float16 v8h;
typedef __attribute__((ext_vector_type(16))) _Float16 v16h;
typedef __attribute__((ext_vector_type(8)))  float    v8f;
typedef __attribute__((ext_vector_type(4)))  unsigned int v4u;
typedef __attribute__((ext_vector_type(8)))  int      v8i;
typedef __attribute__((ext_vector_type(4)))  int      v4i;

#define N_TOK 2048
#define C_DIM 256
#define NHEAD 8
#define HDIM 32
#define BATCH 4
#define W_IMG 64
#define H_IMG 32
#define LN_EPS 1e-5f
#define ATTN_SCALE 0.17677669529663687f /* 32^-0.5 */

#define SHUF16(lo, hi) __builtin_shufflevector(lo, hi, 0,1,2,3,4,5,6,7,8,9,10,11,12,13,14,15)

__device__ __forceinline__ float gelu_exact(float v) {
  return 0.5f * v * (1.0f + erff(v * 0.7071067811865476f));
}

// ---------------------------------------------------------------------------
// TDM: issue a 2D tensor tile DMA  global(f16) -> LDS  (descriptor per
// CDNA5 ISA D# group0/group1 bitfields).
//   tile = tile_d0 (contiguous f16) x tile_d1 rows, row pitch stride0 (f16).
//   LDS receives the tile packed row-major (pitch = tile_d0).
// Tracked by TENSORcnt; caller waits with s_wait_tensorcnt.
// ---------------------------------------------------------------------------
__device__ __forceinline__ void tdm_load_2d(const void* gaddr, unsigned lds_off,
                                            unsigned tensor_d0, unsigned tensor_d1,
                                            unsigned stride0,
                                            unsigned tile_d0, unsigned tile_d1) {
  const unsigned long long ga = (unsigned long long)(size_t)gaddr;
  v4u g0;
  g0[0] = 1u;                                          // count=1, user D#
  g0[1] = lds_off;                                     // lds_addr (bytes)
  g0[2] = (unsigned)(ga & 0xFFFFFFFFu);                // global_addr[31:0]
  g0[3] = (unsigned)((ga >> 32) & 0x1FFFFFFu)          // global_addr[56:32]
        | (2u << 30);                                  // type = 2 ("image")
  v8i g1;
  g1[0] = (int)(1u << 16);                             // data_size=1 -> 2 bytes
  g1[1] = (int)((tensor_d0 & 0xFFFFu) << 16);          // tensor_dim0[15:0]
  g1[2] = (int)((tensor_d0 >> 16) | ((tensor_d1 & 0xFFFFu) << 16));
  g1[3] = (int)((tensor_d1 >> 16) | (tile_d0 << 16));  // tile_dim0
  g1[4] = (int)(tile_d1 & 0xFFFFu);                    // tile_dim1 (tile_dim2=0)
  g1[5] = (int)stride0;                                // tensor_dim0_stride lo32
  g1[6] = 0;
  g1[7] = 0;
  const v4i z4 = {0, 0, 0, 0};
#if __clang_major__ >= 23
  const v8i z8 = {0, 0, 0, 0, 0, 0, 0, 0};
  __builtin_amdgcn_tensor_load_to_lds(g0, g1, z4, z4, z8, 0);
#else
  __builtin_amdgcn_tensor_load_to_lds(g0, g1, z4, z4, 0);
#endif
}

__device__ __forceinline__ unsigned lds_off_of(const void* p) {
  return (unsigned)(size_t)p;   // LDS flat address: low 32 bits = LDS offset
}

// ---------------------------------------------------------------------------
// fp32 -> f16 elementwise convert
// ---------------------------------------------------------------------------
__global__ __launch_bounds__(256) void cvt_f16(const float* __restrict__ s,
                                               f16* __restrict__ d, int n) {
  const int i = blockIdx.x * 256 + threadIdx.x;
  if (i < n) d[i] = (f16)s[i];
}

// ---------------------------------------------------------------------------
// f16 WMMA GEMM with TDM-staged, double-buffered LDS operand slabs.
//   Out = A[M,K](f16) @ W[N,K](f16)^T + bias   (f32 accumulate)
// 8 waves (2x4) -> 32x64 macro-tile. K consumed in 64-wide slabs:
//   wave0 issues tensor_load_to_lds for slab s+1, all waves run 2 WMMAs on
//   slab s from LDS, then s_wait_tensorcnt(0) + barrier flips the buffer.
// ---------------------------------------------------------------------------
__global__ __launch_bounds__(256) void gemm_wmma_f16(
    const f16* __restrict__ A, const f16* __restrict__ W,
    const float* __restrict__ bias, float* __restrict__ outf,
    f16* __restrict__ outh, int M, int N, int K, int act) {
  const int lane = threadIdx.x & 31;
  const int wave = threadIdx.x >> 5;
  const int row  = lane & 15;
  const int hi   = lane >> 4;
  const int wm   = wave >> 2;       // 0..1
  const int wn   = wave & 3;        // 0..3
  const int m0 = blockIdx.x * 32;
  const int n0 = blockIdx.y * 64;

  const int KS = 64;                // k-slab width (f16 elements)
  __shared__ __align__(16) f16 sA[2][32 * 64];   // [row-in-block][k] packed
  __shared__ __align__(16) f16 sW[2][64 * 64];

  const int nslab = K / KS;
  const f16* Ab = A + (size_t)m0 * K;
  const f16* Wb = W + (size_t)n0 * K;

  if (wave == 0) {                  // TDM ignores EXEC; gate by wave
    tdm_load_2d(Ab, lds_off_of(sA[0]), K, 32, K, KS, 32);
    tdm_load_2d(Wb, lds_off_of(sW[0]), K, 64, K, KS, 64);
    __builtin_amdgcn_s_wait_tensorcnt(0);
  }
  __syncthreads();

  v8f acc = {};
  for (int s = 0; s < nslab; ++s) {
    const int cur = s & 1;
    if (wave == 0 && s + 1 < nslab) {       // overlap next slab's DMA
      const int k = (s + 1) * KS;
      tdm_load_2d(Ab + k, lds_off_of(sA[cur ^ 1]), K, 32, K, KS, 32);
      tdm_load_2d(Wb + k, lds_off_of(sW[cur ^ 1]), K, 64, K, KS, 64);
    }
    const f16* a0 = sA[cur] + (wm * 16 + row) * KS;
    const f16* w0 = sW[cur] + (wn * 16 + row) * KS;
    #pragma unroll
    for (int kk = 0; kk < KS; kk += 32) {
      v16h av = *(const v16h*)(a0 + kk + 16 * hi);
      v16h bv = *(const v16h*)(w0 + kk + 16 * hi);
      acc = __builtin_amdgcn_wmma_f32_16x16x32_f16(false, av, false, bv,
                                                   (short)0, acc, false, false);
    }
    __syncthreads();                        // all waves done with slab `cur`
    if (wave == 0 && s + 1 < nslab)
      __builtin_amdgcn_s_wait_tensorcnt(0); // next slab landed in LDS
    __syncthreads();
  }

  const int c = n0 + wn * 16 + row;
  const int mb = m0 + wm * 16;
  const float bvs = bias[c];
  #pragma unroll
  for (int j = 0; j < 8; ++j) {             // D: vgpr j -> rows j / j+8
    const int r = mb + j + 8 * hi;
    float v = acc[j] + bvs;
    if (act) v = gelu_exact(v);
    if (outf) outf[(size_t)r * N + c] = v;
    if (outh) outh[(size_t)r * N + c] = (f16)v;
  }
}

// ---------------------------------------------------------------------------
// Local attention (flash-style), f16 operands, f32 softmax/accumulators.
// Mask analytic: key (jj,ii) valid for query (j,i) iff |jj-j|<=3 && |ii-i|<=5.
// Block = (image row, head, batch): 64 queries x <=448 contiguous key tokens.
// 4 waves, wave = 16-query tile. 32 keys/iteration: S 2 WMMAs, PV 2 WMMAs.
// ---------------------------------------------------------------------------
__global__ __launch_bounds__(128) void local_attn(
    const f16* __restrict__ qkv, f16* __restrict__ out) {
  const int jrow = blockIdx.x;
  const int h    = blockIdx.y;
  const int b    = blockIdx.z;
  const int lane = threadIdx.x & 31;
  const int wave = threadIdx.x >> 5;
  const int row  = lane & 15;
  const int hi   = lane >> 4;

  __shared__ __align__(16) f16 sK [32 * 48];     // [key][d]  pitch 48 (96B)
  __shared__ __align__(16) f16 sVt[32 * 48];     // [d][key]  pitch 48
  __shared__ __align__(16) f16 sP [4 * 16 * 40]; // per-wave [qrow][key]
  f16* pb = sP + wave * 16 * 40;

  const int jr0   = (jrow - 3 < 0) ? 0 : jrow - 3;
  const int jr1   = (jrow + 3 > H_IMG - 1) ? H_IMG - 1 : jrow + 3;
  const int nkeys = (jr1 - jr0 + 1) * W_IMG;     // contiguous tokens
  const int nkt   = nkeys >> 5;
  const int ktok0 = jr0 * W_IMG;

  const int qtok = jrow * W_IMG + wave * 16 + row;
  const f16* qp = qkv + (size_t)(b * N_TOK + qtok) * 768 + h * HDIM;
  v16h aq = *(const v16h*)(qp + 16 * hi);

  float mrow[8], lrow[8];
  v8f o1 = {}, o2 = {};
  #pragma unroll
  for (int j = 0; j < 8; ++j) { mrow[j] = -1e30f; lrow[j] = 0.0f; }

  for (int kt = 0; kt < nkt; ++kt) {
    __syncthreads();
    {   // cooperative load: 32 keys x 32 dims of K and V (V transposed)
      const int t  = threadIdx.x;
      const int kr = t >> 2;
      const int c8 = (t & 3) * 8;
      const int ktok = ktok0 + kt * 32 + kr;
      const f16* kp = qkv + (size_t)(b * N_TOK + ktok) * 768 + 256 + h * HDIM + c8;
      if (kt + 1 < nkt) __builtin_prefetch(kp + 32 * 768, 0, 1);
      *(v8h*)(sK + kr * 48 + c8) = *(const v8h*)kp;
      v8h vv = *(const v8h*)(kp + 256);
      #pragma unroll
      for (int e = 0; e < 8; ++e) sVt[(c8 + e) * 48 + kr] = vv[e];
    }
    __syncthreads();

    v16h bk0 = *(const v16h*)(sK + row * 48 + 16 * hi);
    v16h bk1 = *(const v16h*)(sK + (16 + row) * 48 + 16 * hi);
    v8f s0 = {}, s1 = {};
    s0 = __builtin_amdgcn_wmma_f32_16x16x32_f16(false, aq, false, bk0,
                                                (short)0, s0, false, false);
    s1 = __builtin_amdgcn_wmma_f32_16x16x32_f16(false, aq, false, bk1,
                                                (short)0, s1, false, false);

    const int kcol0 = ((kt & 1) << 5) + row;
    const int kcol1 = kcol0 + 16;
    #pragma unroll
    for (int j = 0; j < 8; ++j) {
      const int qi = wave * 16 + j + 8 * hi;
      const int d0 = kcol0 - qi, d1 = kcol1 - qi;
      float sv0 = s0[j] * ATTN_SCALE + ((d0 <= 5 && d0 >= -5) ? 0.0f : -1e30f);
      float sv1 = s1[j] * ATTN_SCALE + ((d1 <= 5 && d1 >= -5) ? 0.0f : -1e30f);
      float tm = fmaxf(sv0, sv1);
      tm = fmaxf(tm, __shfl_xor(tm, 1));
      tm = fmaxf(tm, __shfl_xor(tm, 2));
      tm = fmaxf(tm, __shfl_xor(tm, 4));
      tm = fmaxf(tm, __shfl_xor(tm, 8));
      const float mnew = fmaxf(mrow[j], tm);
      const float p0 = __expf(sv0 - mnew);
      const float p1 = __expf(sv1 - mnew);
      float ts = p0 + p1;
      ts += __shfl_xor(ts, 1);
      ts += __shfl_xor(ts, 2);
      ts += __shfl_xor(ts, 4);
      ts += __shfl_xor(ts, 8);
      const float sc = __expf(mrow[j] - mnew);
      lrow[j] = lrow[j] * sc + ts;
      o1[j] *= sc;
      o2[j] *= sc;
      mrow[j] = mnew;
      pb[(j + 8 * hi) * 40 + row]      = (f16)p0;
      pb[(j + 8 * hi) * 40 + 16 + row] = (f16)p1;
    }
    __syncthreads();

    v8h plo = *(const v8h*)(pb + row * 40 + 16 * hi);
    v8h phi = *(const v8h*)(pb + row * 40 + 16 * hi + 8);
    v16h ap = SHUF16(plo, phi);
    v16h bv0 = *(const v16h*)(sVt + row * 48 + 16 * hi);
    v16h bv1 = *(const v16h*)(sVt + (16 + row) * 48 + 16 * hi);
    o1 = __builtin_amdgcn_wmma_f32_16x16x32_f16(false, ap, false, bv0,
                                                (short)0, o1, false, false);
    o2 = __builtin_amdgcn_wmma_f32_16x16x32_f16(false, ap, false, bv1,
                                                (short)0, o2, false, false);
  }

  #pragma unroll
  for (int j = 0; j < 8; ++j) {
    const int tok = jrow * W_IMG + wave * 16 + j + 8 * hi;
    const float inv = 1.0f / lrow[j];
    f16* op = out + (size_t)(b * N_TOK + tok) * C_DIM + h * HDIM;
    op[row]      = (f16)(o1[j] * inv);
    op[16 + row] = (f16)(o2[j] * inv);
  }
}

// ---------------------------------------------------------------------------
// Fused residual + LayerNorm (C=256), optional extra f16 copy of the output.
// ---------------------------------------------------------------------------
__global__ __launch_bounds__(256) void resid_ln(
    const float* __restrict__ x, const float* __restrict__ r,
    const float* __restrict__ g, const float* __restrict__ bta,
    float* __restrict__ out, f16* __restrict__ outh) {
  const int rowid = blockIdx.x;
  const int t = threadIdx.x;
  const size_t base = (size_t)rowid * C_DIM;
  const float v = x[base + t] + r[base + t];

  __shared__ float red[256];
  red[t] = v;
  __syncthreads();
  for (int off = 128; off > 0; off >>= 1) {
    if (t < off) red[t] += red[t + off];
    __syncthreads();
  }
  const float mu = red[0] * (1.0f / 256.0f);
  __syncthreads();
  const float dv = v - mu;
  red[t] = dv * dv;
  __syncthreads();
  for (int off = 128; off > 0; off >>= 1) {
    if (t < off) red[t] += red[t + off];
    __syncthreads();
  }
  const float var = red[0] * (1.0f / 256.0f);
  const float y = dv * rsqrtf(var + LN_EPS) * g[t] + bta[t];
  out[base + t] = y;
  if (outh) outh[base + t] = (f16)y;
}

// ---------------------------------------------------------------------------
extern "C" void kernel_launch(void* const* d_in, const int* in_sizes, int n_in,
                              void* d_out, int out_size, void* d_ws, size_t ws_size,
                              hipStream_t stream) {
  const float* x      = (const float*)d_in[0];
  /* d_in[1] = mask: unused — local window recomputed analytically */
  const float* qkv_w  = (const float*)d_in[2];
  const float* qkv_b  = (const float*)d_in[3];
  const float* proj_w = (const float*)d_in[4];
  const float* proj_b = (const float*)d_in[5];
  const float* ln1_g  = (const float*)d_in[6];
  const float* ln1_b  = (const float*)d_in[7];
  const float* w1     = (const float*)d_in[8];
  const float* b1     = (const float*)d_in[9];
  const float* w2     = (const float*)d_in[10];
  const float* b2     = (const float*)d_in[11];
  const float* ln2_g  = (const float*)d_in[12];
  const float* ln2_b  = (const float*)d_in[13];
  float* outp = (float*)d_out;

  const int M = BATCH * N_TOK;  // 8192

  float* tmp = (float*)d_ws;                 // [M,256] f32
  float* x1  = tmp + (size_t)M * 256;        // [M,256] f32
  f16* x16    = (f16*)(x1 + (size_t)M * 256);
  f16* qkv16  = x16   + (size_t)M * 256;     // [M,768]
  f16* attn16 = qkv16 + (size_t)M * 768;     // [M,256]
  f16* x1h    = attn16 + (size_t)M * 256;    // [M,256]
  f16* h16    = x1h   + (size_t)M * 256;     // [M,1024]
  f16* wq16   = h16   + (size_t)M * 1024;    // [768,256]
  f16* wp16   = wq16  + 768 * 256;           // [256,256]
  f16* w116   = wp16  + 256 * 256;           // [1024,256]
  f16* w216   = w116  + 1024 * 256;          // [256,1024]

  cvt_f16<<<(M * 256 + 255) / 256, 256, 0, stream>>>(x, x16, M * 256);
  cvt_f16<<<(768 * 256 + 255) / 256, 256, 0, stream>>>(qkv_w, wq16, 768 * 256);
  cvt_f16<<<(256 * 256 + 255) / 256, 256, 0, stream>>>(proj_w, wp16, 256 * 256);
  cvt_f16<<<(1024 * 256 + 255) / 256, 256, 0, stream>>>(w1, w116, 1024 * 256);
  cvt_f16<<<(1024 * 256 + 255) / 256, 256, 0, stream>>>(w2, w216, 1024 * 256);

  gemm_wmma_f16<<<dim3(M / 32, 768 / 64), 256, 0, stream>>>(
      x16, wq16, qkv_b, nullptr, qkv16, M, 768, 256, 0);
  local_attn<<<dim3(H_IMG, NHEAD, BATCH), 128, 0, stream>>>(qkv16, attn16);
  gemm_wmma_f16<<<dim3(M / 32, 256 / 64), 256, 0, stream>>>(
      attn16, wp16, proj_b, tmp, nullptr, M, 256, 256, 0);
  resid_ln<<<M, 256, 0, stream>>>(x, tmp, ln1_g, ln1_b, x1, x1h);
  gemm_wmma_f16<<<dim3(M / 32, 1024 / 64), 256, 0, stream>>>(
      x1h, w116, b1, nullptr, h16, M, 1024, 256, 1);
  gemm_wmma_f16<<<dim3(M / 32, 256 / 64), 256, 0, stream>>>(
      h16, w216, b2, tmp, nullptr, M, 256, 1024, 0);
  resid_ln<<<M, 256, 0, stream>>>(x1, tmp, ln2_g, ln2_b, outp, nullptr);
}
